// FastUpConvolution_7378753814848
// MI455X (gfx1250) — matrix operand order repaired and checked by reading, not compiled, not run on hardware
//
#include <hip/hip_runtime.h>
#include <hip/hip_bf16.h>
#include <stdint.h>

// ---------------------------------------------------------------------------
// FastUpConvolution for MI455X (gfx1250): 4-branch conv as implicit GEMM on
// v_wmma_f32_16x16x32_bf16, pixel-shuffle epilogue, BN(training)+ReLU.
// Wave tile M=32 x N=32: each B fragment feeds 2 WMMAs (halves L2 weight BW).
// ---------------------------------------------------------------------------

typedef float  v8f   __attribute__((ext_vector_type(8)));
typedef __bf16 v16bf __attribute__((ext_vector_type(16)));

union Frag32 { uint4 u[2]; v16bf v; };   // 32 bytes = 16 bf16 per lane

__device__ __forceinline__ unsigned short f2bf(float f) {
  unsigned int u = __float_as_uint(f);
  u += 0x7FFFu + ((u >> 16) & 1u);       // round-to-nearest-even
  return (unsigned short)(u >> 16);
}

// ---- prep: x [n][ci][h][w] f32  ->  xT [n][h][w][ci] bf16 (ci contiguous) --
__global__ void k_prep_x(const float* __restrict__ x, unsigned short* __restrict__ xT) {
  const int pix = blockIdx.x * 256 + threadIdx.x;          // 16384 pixels
  const int w = pix & 31, h = (pix >> 5) & 31, nn = pix >> 10;
  const float* src = x + (size_t)nn * (512 * 1024) + h * 32 + w;
  unsigned int* dst = (unsigned int*)(xT + (size_t)pix * 512);
#pragma unroll 4
  for (int ci = 0; ci < 512; ci += 2) {
    float a = src[(size_t)ci * 1024];
    float b = src[(size_t)(ci + 1) * 1024];
    dst[ci >> 1] = (unsigned int)f2bf(a) | ((unsigned int)f2bf(b) << 16);
  }
}

// ---- prep: w1..w4 f32 -> Wb [25 slots][cout][cin] bf16 ---------------------
// slot order: kh-major, kw, then active branches {o1 always, o2 kh<2, o3 kw<2, o4 both}
__device__ const int g_slot_br[25] = {0,1,2,3, 0,1,2,3, 0,1, 0,1,2,3, 0,1,2,3, 0,1, 0,2, 0,2, 0};
__device__ const int g_slot_kh[25] = {0,0,0,0, 0,0,0,0, 0,0, 1,1,1,1, 1,1,1,1, 1,1, 2,2, 2,2, 2};
__device__ const int g_slot_kw[25] = {0,0,0,0, 1,1,1,1, 2,2, 0,0,0,0, 1,1,1,1, 2,2, 0,0, 1,1, 2};
__device__ const int g_khd[4] = {3,2,3,2};
__device__ const int g_kwd[4] = {3,3,2,2};

__global__ void k_prep_w(const float* __restrict__ w1, const float* __restrict__ w2,
                         const float* __restrict__ w3, const float* __restrict__ w4,
                         unsigned short* __restrict__ Wb) {
  const int idx = blockIdx.x * 256 + threadIdx.x;          // 25*256*512 = 3276800
  const int ci = idx & 511;
  const int co = (idx >> 9) & 255;
  const int slot = idx >> 17;
  const int br = g_slot_br[slot], kh = g_slot_kh[slot], kw = g_slot_kw[slot];
  const float* wp = (br == 0) ? w1 : (br == 1) ? w2 : (br == 2) ? w3 : w4;
  const int khd = g_khd[br], kwd = g_kwd[br];
  const float v = wp[((size_t)(co * 512 + ci) * khd + kh) * kwd + kw];
  Wb[idx] = f2bf(v);
}

// ---- main conv: WMMA implicit GEMM ----------------------------------------
// Block: 512 threads = 16 waves (wm 0..1 rows x wn 0..7 cout-slices).
// Block tile: 2 input rows x 32 cols x 256 couts x 4 branches.
// Wave tile: M=32 pixels (one row), N=32 couts; acc = 4 br x 2 mi x 2 ns v8f.
__launch_bounds__(512)
__global__ void k_conv_wmma(const unsigned short* __restrict__ xT,
                            const unsigned short* __restrict__ Wb,
                            const float* __restrict__ b1, const float* __restrict__ b2,
                            const float* __restrict__ b3, const float* __restrict__ b4,
                            float* __restrict__ y) {
  __shared__ __align__(16) unsigned short xs[4 * 34 * 40];  // rows -1..+2, cols -1..32, 32 ci, stride 40
  __shared__ float2 yst[64 * 66];                           // store-transpose buffer (64 px x 64 co)

  const int tid  = threadIdx.x;
  const int lane = tid & 31;
  const int wave = tid >> 5;
  const int wm = wave & 1;          // which of the 2 input rows
  const int wn = wave >> 1;         // 32-cout slice 0..7
  const int ln = lane & 15;
  const int hiL = (lane >> 4) & 1;
  const int aLo = hiL ? 8 : 0;      // A-frag K offset (ISA 16-bit A 16x32 layout)
  const int bLo = hiL ? 8 : 0;      // B-frag K offset (matches A-frag K order)

  const int n   = blockIdx.x >> 4;
  const int hr0 = (blockIdx.x & 15) << 1;

  v8f acc[4][2][2];                 // [branch][mi (w-half)][ns]
  {
    v8f z = {0.f,0.f,0.f,0.f,0.f,0.f,0.f,0.f};
#pragma unroll
    for (int i = 0; i < 4; ++i)
#pragma unroll
      for (int mi = 0; mi < 2; ++mi)
#pragma unroll
        for (int j = 0; j < 2; ++j) acc[i][mi][j] = z;
  }

  constexpr int tap_nbr[9] = {4,4,2, 4,4,2, 2,2,1};
  constexpr int tap_s0[9]  = {0,4,8, 10,14,18, 20,22,24};
  constexpr int tap_br[9][4] = {{0,1,2,3},{0,1,2,3},{0,1,0,0},
                                {0,1,2,3},{0,1,2,3},{0,1,0,0},
                                {0,2,0,0},{0,2,0,0},{0,0,0,0}};

  for (int kc = 0; kc < 16; ++kc) {           // 16 chunks of 32 input channels
    __syncthreads();
    // prefetch next chunk's x rows (gfx1250 global_prefetch_b8)
    if (kc < 15)
      __builtin_prefetch(xT + ((size_t)((n * 32 + hr0) * 32) << 9) + ((kc + 1) << 5), 0, 1);
    // stage X chunk (with halo + zero pad) into LDS, bf16, ci-contiguous
    for (int idx = tid; idx < 4 * 34 * 32; idx += 512) {
      const int r   = idx / (34 * 32);
      const int rem = idx - r * (34 * 32);
      const int c   = rem >> 5;
      const int ci  = rem & 31;
      const int rg  = hr0 - 1 + r;
      const int wg  = c - 1;
      unsigned short v = 0;
      if ((unsigned)rg < 32u && (unsigned)wg < 32u)
        v = xT[((size_t)((n * 32 + rg) * 32 + wg) << 9) + (kc << 5) + ci];
      xs[(r * 34 + c) * 40 + ci] = v;
    }
    __syncthreads();

#pragma unroll
    for (int t = 0; t < 9; ++t) {             // 9 global taps of the 3x3 window
      const int kh = t / 3, kw = t % 3;
      const int row = wm + kh;                              // LDS row index 0..3
      Frag32 A[2];
#pragma unroll
      for (int mi = 0; mi < 2; ++mi) {
        const int col = (mi << 4) + ln + kw;                // LDS col index 0..33
        const unsigned short* ap = &xs[(row * 34 + col) * 40 + aLo];
        A[mi].u[0] = *(const uint4*)(ap);                   // K = aLo..aLo+7
        A[mi].u[1] = *(const uint4*)(ap + 16);              // K = aLo+16..aLo+23
      }
#pragma unroll
      for (int j = 0; j < tap_nbr[t]; ++j) {                // branches sharing this tap
        const int br   = tap_br[t][j];
        const int slot = tap_s0[t] + j;
#pragma unroll
        for (int ns = 0; ns < 2; ++ns) {
          const int co = (wn << 5) + (ns << 4) + ln;
          const unsigned short* bp =
              Wb + (((size_t)(slot * 256 + co)) << 9) + (kc << 5) + bLo;
          Frag32 B;
          B.u[0] = *(const uint4*)(bp);                     // K = bLo..bLo+7
          B.u[1] = *(const uint4*)(bp + 16);                // K = bLo+16..bLo+23
#pragma unroll
          for (int mi = 0; mi < 2; ++mi) {                  // B reused for both w-halves
            acc[br][mi][ns] = __builtin_amdgcn_wmma_f32_16x16x32_bf16(
                false, A[mi].v, false, B.v, (short)0, acc[br][mi][ns], false, false);
          }
        }
      }
    }
  }

  // ---- epilogue: +bias, pixel-shuffle interleave, coalesced float2 stores --
  // y[2h+pr][2w+pc]: br0->(0,0) br1->(1,0) br2->(0,1) br3->(1,1)
  const int mh = hiL ? 8 : 0;
#pragma unroll
  for (int pr = 0; pr < 2; ++pr) {
    const float* bAp = pr ? b2 : b1;   // pc=0 branch for this row parity
    const float* bBp = pr ? b4 : b3;   // pc=1 branch
    const int brA = pr, brB = pr + 2;
    for (int g = 0; g < 4; ++g) {      // 64-cout groups
      __syncthreads();
      if ((wn >> 1) == g) {            // the 4 waves owning this cout group
#pragma unroll
        for (int ns = 0; ns < 2; ++ns) {
          const int col = ((wn & 1) << 5) + (ns << 4) + ln;   // 0..63 within group
          const float bA = bAp[(g << 6) + col];
          const float bB = bBp[(g << 6) + col];
#pragma unroll
          for (int mi = 0; mi < 2; ++mi) {
#pragma unroll
            for (int v = 0; v < 8; ++v) {
              const int p = (wm << 5) + (mi << 4) + v + mh;   // 0..63
              yst[p * 66 + col] = make_float2(acc[brA][mi][ns][v] + bA,
                                              acc[brB][mi][ns][v] + bB);
            }
          }
        }
      }
      __syncthreads();
      for (int idx = tid; idx < 4096; idx += 512) {       // 64 px x 64 co float2
        const int col = idx >> 6;
        const int p   = idx & 63;
        const int r = p >> 5, w = p & 31;
        const int oh = ((hr0 + r) << 1) + pr;
        const int co = (g << 6) + col;
        float2 val = yst[p * 66 + col];
        *(float2*)(y + (((size_t)(n * 256 + co) * 64 + oh) * 64 + (w << 1))) = val;
      }
    }
  }
}

// ---- BN stats: per-channel scale/shift ------------------------------------
__global__ void k_stats(const float* __restrict__ y, const float* __restrict__ gamma,
                        const float* __restrict__ beta, float* __restrict__ stats) {
  const int co = blockIdx.x;
  const int tid = threadIdx.x;
  float s = 0.f, s2 = 0.f;
  for (int nn = 0; nn < 16; ++nn) {
    const float4* p = (const float4*)(y + ((size_t)(nn * 256 + co) << 12));
    for (int i = tid; i < 1024; i += 256) {
      float4 v = p[i];
      s  += v.x + v.y + v.z + v.w;
      s2 += v.x * v.x + v.y * v.y + v.z * v.z + v.w * v.w;
    }
  }
  __shared__ float rs[256], rq[256];
  rs[tid] = s; rq[tid] = s2;
  __syncthreads();
  for (int off = 128; off > 0; off >>= 1) {
    if (tid < off) { rs[tid] += rs[tid + off]; rq[tid] += rq[tid + off]; }
    __syncthreads();
  }
  if (tid == 0) {
    const float inv = 1.f / 65536.f;
    const float mean = rs[0] * inv;
    const float var  = rq[0] * inv - mean * mean;
    const float sc = gamma[co] * rsqrtf(var + 1e-5f);
    stats[co]       = sc;
    stats[256 + co] = beta[co] - mean * sc;
  }
}

// ---- in-place normalize + ReLU --------------------------------------------
__global__ void k_norm(float* __restrict__ y, const float* __restrict__ stats) {
  const size_t i = (size_t)blockIdx.x * 256 + threadIdx.x;  // float4 index, 4194304 total
  const int co = (int)((i >> 10) & 255);
  const float sc = stats[co], sh = stats[256 + co];
  float4* p = (float4*)y;
  float4 v = p[i];
  v.x = fmaxf(fmaf(v.x, sc, sh), 0.f);
  v.y = fmaxf(fmaf(v.y, sc, sh), 0.f);
  v.z = fmaxf(fmaf(v.z, sc, sh), 0.f);
  v.w = fmaxf(fmaf(v.w, sc, sh), 0.f);
  p[i] = v;
}

// ---------------------------------------------------------------------------
extern "C" void kernel_launch(void* const* d_in, const int* in_sizes, int n_in,
                              void* d_out, int out_size, void* d_ws, size_t ws_size,
                              hipStream_t stream) {
  const float* x  = (const float*)d_in[0];
  const float* w1 = (const float*)d_in[1];
  const float* b1 = (const float*)d_in[2];
  const float* w2 = (const float*)d_in[3];
  const float* b2 = (const float*)d_in[4];
  const float* w3 = (const float*)d_in[5];
  const float* b3 = (const float*)d_in[6];
  const float* w4 = (const float*)d_in[7];
  const float* b4 = (const float*)d_in[8];
  const float* gamma = (const float*)d_in[9];
  const float* beta  = (const float*)d_in[10];
  float* y = (float*)d_out;

  // workspace: xT (16 MiB bf16) | Wb (6.25 MiB bf16) | stats (2 KiB f32)
  unsigned short* xT = (unsigned short*)d_ws;
  unsigned short* Wb = (unsigned short*)((char*)d_ws + 16777216);
  float* stats = (float*)((char*)d_ws + 16777216 + 6553600);

  k_prep_x<<<64, 256, 0, stream>>>(x, xT);
  k_prep_w<<<12800, 256, 0, stream>>>(w1, w2, w3, w4, Wb);
  k_conv_wmma<<<256, 512, 0, stream>>>(xT, Wb, b1, b2, b3, b4, y);
  k_stats<<<256, 256, 0, stream>>>(y, gamma, beta, stats);
  k_norm<<<16384, 256, 0, stream>>>(y, stats);
}
